// DynamicsTransformer_89163521065230
// MI455X (gfx1250) — matrix operand order repaired and checked by reading, not compiled, not run on hardware
//
#include <hip/hip_runtime.h>
#include <math.h>

// ---------------- model constants ----------------
constexpr int B_   = 2;
constexpr int T_   = 2048;
constexpr int D_   = 2048;
constexpr int H_   = 16;
constexpr int KVH_ = 4;
constexpr int HD_  = 128;
constexpr float CAP_ = 50.0f;
constexpr float SCALE_ = 0.08838834764831845f; // 128^-0.5

typedef __attribute__((ext_vector_type(16))) __bf16 v16bf;
typedef __attribute__((ext_vector_type(8)))  float  v8f;
typedef __attribute__((ext_vector_type(4)))  unsigned u32x4;
typedef __attribute__((ext_vector_type(8)))  int      i32x8;
typedef __attribute__((ext_vector_type(4)))  int      i32x4;

union BF16x16 { v16bf v; unsigned u[8]; };

__device__ __forceinline__ unsigned short f2bf(float f) {
  unsigned u = __float_as_uint(f);
  unsigned r = 0x7FFFu + ((u >> 16) & 1u);
  return (unsigned short)((u + r) >> 16);
}

// A-fragment (16x32 bf16) k-offset for dword j, lane-half hi
__device__ __forceinline__ int a_koff(int j, int hi) {
  return (j < 4 ? 2 * j : 2 * j + 8) + 8 * hi;
}

// ---------------- TDM: 2D tile load (global -> LDS) ----------------
// Elements are handled in 4-byte units (data_size=2 -> 4B).
// tile_dw x tile_rows tile; row stride row_stride_dw dwords; tensor dims for
// OOB are set to the full tensor so no clipping occurs.
__device__ __forceinline__ void tdm_load_2d(const void* lds_ptr,
                                            const void* gptr,
                                            unsigned tile_rows,
                                            unsigned tile_dw,
                                            unsigned row_stride_dw,
                                            unsigned tensor_dim0_dw,
                                            unsigned tensor_dim1_rows) {
  unsigned long long ga = (unsigned long long)(uintptr_t)gptr;
  unsigned lds_off = (unsigned)(unsigned long long)(uintptr_t)lds_ptr;
  u32x4 g0;
  g0[0] = 1u;                                   // count=1 (valid), user mode
  g0[1] = lds_off;                              // lds_addr [63:32]
  g0[2] = (unsigned)(ga & 0xFFFFFFFFu);         // global_addr [95:64]
  g0[3] = (unsigned)((ga >> 32) & 0x01FFFFFFu)  // global_addr [120:96]
          | (2u << 30);                         // type=2 ("image") [127:126]
  i32x8 g1;
  g1[0] = (int)(2u << 16);                      // wg_mask=0, data_size=2 (4B)
  g1[1] = (int)((tensor_dim0_dw & 0xFFFFu) << 16);            // dim0[15:0]
  g1[2] = (int)((tensor_dim0_dw >> 16) & 0xFFFFu)             // dim0[31:16]
          | (int)((tensor_dim1_rows & 0xFFFFu) << 16);        // dim1[15:0]
  g1[3] = (int)((tensor_dim1_rows >> 16) & 0xFFFFu)           // dim1[31:16]
          | (int)((tile_dw & 0xFFFFu) << 16);                 // tile_dim0
  g1[4] = (int)(tile_rows & 0xFFFFu);                         // tile_dim1, tile_dim2=0
  g1[5] = (int)row_stride_dw;                                 // dim0_stride[31:0]
  g1[6] = 0;                                                  // stride hi / dim1_stride lo
  g1[7] = 0;                                                  // dim1_stride hi
  i32x4 z;
  z[0] = 0; z[1] = 0; z[2] = 0; z[3] = 0;
  i32x8 z8;
  z8[0] = 0; z8[1] = 0; z8[2] = 0; z8[3] = 0;
  z8[4] = 0; z8[5] = 0; z8[6] = 0; z8[7] = 0;
  __builtin_amdgcn_tensor_load_to_lds(g0, g1, z, z, z8, 0);
}

// ---------------- f32 -> bf16 convert ----------------
__global__ void cvt_kernel(const float* __restrict__ in,
                           unsigned short* __restrict__ out, int n) {
  for (int i = blockIdx.x * blockDim.x + threadIdx.x; i < n;
       i += gridDim.x * blockDim.x)
    out[i] = f2bf(in[i]);
}

// ---------------- TDM-fed WMMA GEMM: C(M,N) = A(M,K) * W(N,K)^T ----------------
// A row-major bf16, W row-major bf16 (N rows of K), C row-major f32.
// block = 256 threads (8 waves) computes a 128x128 C tile; wave w owns rows
// 16*w..16*w+15 and all 128 columns. A/B 128x32 tiles staged in LDS by the
// Tensor Data Mover with double buffering (TENSORcnt).
__global__ __launch_bounds__(256) void gemm_bf16_kernel(
    const unsigned short* __restrict__ A, const unsigned short* __restrict__ W,
    float* __restrict__ C, int M, int N, int K) {
  __shared__ unsigned short Abuf[2][128 * 32];
  __shared__ unsigned short Bbuf[2][128 * 32];

  const int lane = threadIdx.x & 31;
  const int wave = threadIdx.x >> 5;
  const int ln = lane & 15;
  const int hi = lane >> 4;
  const int mblk = blockIdx.y * 128;
  const int nblk = blockIdx.x * 128;
  const int nk = K / 32;
  const unsigned kdw = (unsigned)(K / 2);

  v8f c[8];
#pragma unroll
  for (int t = 0; t < 8; ++t)
#pragma unroll
    for (int r = 0; r < 8; ++r) c[t][r] = 0.0f;

  // prologue: stage chunk 0 into buffer 0
  if (wave == 0) {
    tdm_load_2d(&Abuf[0][0], A + (size_t)mblk * K, 128, 16, kdw, kdw,
                (unsigned)M);
    tdm_load_2d(&Bbuf[0][0], W + (size_t)nblk * K, 128, 16, kdw, kdw,
                (unsigned)N);
  }

  for (int kc = 0; kc < nk; ++kc) {
    if (wave == 0) {
      if (kc + 1 < nk) {
        const int nb = (kc + 1) & 1;
        tdm_load_2d(&Abuf[nb][0], A + (size_t)mblk * K + (kc + 1) * 32, 128,
                    16, kdw, kdw, (unsigned)M);
        tdm_load_2d(&Bbuf[nb][0], W + (size_t)nblk * K + (kc + 1) * 32, 128,
                    16, kdw, kdw, (unsigned)N);
        __builtin_amdgcn_s_wait_tensorcnt(2);  // retire current chunk's pair
      } else {
        __builtin_amdgcn_s_wait_tensorcnt(0);
      }
    }
    __syncthreads();
    const unsigned short* At = &Abuf[kc & 1][0];
    const unsigned short* Bt = &Bbuf[kc & 1][0];
    BF16x16 a;
#pragma unroll
    for (int j = 0; j < 8; ++j)
      a.u[j] = *(const unsigned*)(At + (wave * 16 + ln) * 32 + a_koff(j, hi));
#pragma unroll
    for (int t = 0; t < 8; ++t) {
      BF16x16 bb;
#pragma unroll
      for (int j = 0; j < 8; ++j)
        bb.u[j] = *(const unsigned*)(Bt + (t * 16 + ln) * 32 + a_koff(j, hi));
      c[t] = __builtin_amdgcn_wmma_f32_16x16x32_bf16(
          false, a.v, false, bb.v, (short)0, c[t], false, false);
    }
    __syncthreads();
  }

#pragma unroll
  for (int t = 0; t < 8; ++t)
#pragma unroll
    for (int r = 0; r < 8; ++r)
      C[(size_t)(mblk + wave * 16 + r + 8 * hi) * N + nblk + t * 16 + ln] =
          c[t][r];
}

// ---------------- per-head RMSNorm (+optional scale), f32 -> bf16 ----------------
// in: (B*T, NH, HD) f32 ; out: (B, NH, T, HD) bf16. one wave per row.
__global__ __launch_bounds__(256) void rmsnorm_head_kernel(
    const float* __restrict__ in, const float* __restrict__ gamma,
    unsigned short* __restrict__ out, int NH, float scale, int rows) {
  int warp = (blockIdx.x * blockDim.x + threadIdx.x) >> 5;
  int lane = threadIdx.x & 31;
  if (warp >= rows) return;
  int h = warp % NH;
  int bt = warp / NH;
  int b = bt / T_;
  int t = bt % T_;
  const float* p = in + (size_t)warp * HD_;
  float v0 = p[lane * 4 + 0], v1 = p[lane * 4 + 1];
  float v2 = p[lane * 4 + 2], v3 = p[lane * 4 + 3];
  float ss = v0 * v0 + v1 * v1 + v2 * v2 + v3 * v3;
  ss += __shfl_xor(ss, 1, 32);
  ss += __shfl_xor(ss, 2, 32);
  ss += __shfl_xor(ss, 4, 32);
  ss += __shfl_xor(ss, 8, 32);
  ss += __shfl_xor(ss, 16, 32);
  float r = rsqrtf(ss * (1.0f / HD_) + 1e-6f) * scale;
  size_t ob = ((size_t)(b * NH + h) * T_ + t) * HD_ + lane * 4;
  out[ob + 0] = f2bf(v0 * r * gamma[lane * 4 + 0]);
  out[ob + 1] = f2bf(v1 * r * gamma[lane * 4 + 1]);
  out[ob + 2] = f2bf(v2 * r * gamma[lane * 4 + 2]);
  out[ob + 3] = f2bf(v3 * r * gamma[lane * 4 + 3]);
}

// ---------------- V transpose: (B*T, KVH*HD) f32 -> (B,KVH,HD,T) bf16 ----------------
__global__ void vtrans_kernel(const float* __restrict__ in,
                              unsigned short* __restrict__ out) {
  const int n = B_ * T_ * KVH_ * HD_;
  for (int i = blockIdx.x * blockDim.x + threadIdx.x; i < n;
       i += gridDim.x * blockDim.x) {
    int hd = i % HD_;
    int kv = (i / HD_) % KVH_;
    int t = (i / (HD_ * KVH_)) % T_;
    int b = i / (HD_ * KVH_ * T_);
    out[((size_t)(b * KVH_ + kv) * HD_ + hd) * T_ + t] = f2bf(in[i]);
  }
}

// ---------------- flash attention (causal, tanh soft-cap, GQA) ----------------
// Qn: (B,H,T,HD) bf16 (pre-scaled by SCALE); Kn: (B,KVH,T,HD) bf16;
// Vt: (B,KVH,HD,T) bf16; Obf: (B*T, H*HD) bf16. one wave per 16-query tile.
__global__ __launch_bounds__(32) void attn_kernel(
    const unsigned short* __restrict__ Qn, const unsigned short* __restrict__ Kn,
    const unsigned short* __restrict__ Vt, unsigned short* __restrict__ Obf) {
  __shared__ float Pbuf[16][32];
  const int lane = threadIdx.x & 31;
  const int ln = lane & 15;
  const int hi = lane >> 4;
  const int q0 = blockIdx.x * 16;
  const int h = blockIdx.y;
  const int b = blockIdx.z;
  const int kv = h / (H_ / KVH_);

  const unsigned short* qptr = Qn + (size_t)(b * H_ + h) * T_ * HD_;
  const unsigned short* kptr = Kn + (size_t)(b * KVH_ + kv) * T_ * HD_;
  const unsigned short* vptr = Vt + (size_t)(b * KVH_ + kv) * HD_ * T_;

  // Q tile: 4 A-fragments covering HD=128
  BF16x16 qa[4];
#pragma unroll
  for (int hc = 0; hc < 4; ++hc)
#pragma unroll
    for (int j = 0; j < 8; ++j)
      qa[hc].u[j] = *(const unsigned*)(qptr + (size_t)(q0 + ln) * HD_ +
                                       hc * 32 + a_koff(j, hi));

  v8f o[8];
#pragma unroll
  for (int nc = 0; nc < 8; ++nc)
#pragma unroll
    for (int r = 0; r < 8; ++r) o[nc][r] = 0.0f;
  float mrow[8], lrow[8];
#pragma unroll
  for (int r = 0; r < 8; ++r) { mrow[r] = -1e30f; lrow[r] = 0.0f; }

  for (int s0 = 0; s0 <= q0 + 15; s0 += 32) {
    v8f sA, sB;
#pragma unroll
    for (int r = 0; r < 8; ++r) { sA[r] = 0.0f; sB[r] = 0.0f; }
#pragma unroll
    for (int hc = 0; hc < 4; ++hc) {
      BF16x16 kb;
#pragma unroll
      for (int j = 0; j < 8; ++j)
        kb.u[j] = *(const unsigned*)(kptr + (size_t)(s0 + ln) * HD_ + hc * 32 +
                                     a_koff(j, hi));
      sA = __builtin_amdgcn_wmma_f32_16x16x32_bf16(
          false, qa[hc].v, false, kb.v, (short)0, sA, false, false);
    }
#pragma unroll
    for (int hc = 0; hc < 4; ++hc) {
      BF16x16 kb;
#pragma unroll
      for (int j = 0; j < 8; ++j)
        kb.u[j] = *(const unsigned*)(kptr + (size_t)(s0 + 16 + ln) * HD_ +
                                     hc * 32 + a_koff(j, hi));
      sB = __builtin_amdgcn_wmma_f32_16x16x32_bf16(
          false, qa[hc].v, false, kb.v, (short)0, sB, false, false);
    }
    // soft-cap, causal mask, online softmax
    const int qbase = q0 + 8 * hi;
#pragma unroll
    for (int r = 0; r < 8; ++r) {
      const int qrow = qbase + r;
      float a = CAP_ * tanhf(sA[r] * (1.0f / CAP_));
      float cc = CAP_ * tanhf(sB[r] * (1.0f / CAP_));
      a = ((s0 + ln) > qrow) ? -1e30f : a;
      cc = ((s0 + 16 + ln) > qrow) ? -1e30f : cc;
      float rm = fmaxf(a, cc);
      rm = fmaxf(rm, __shfl_xor(rm, 1, 32));
      rm = fmaxf(rm, __shfl_xor(rm, 2, 32));
      rm = fmaxf(rm, __shfl_xor(rm, 4, 32));
      rm = fmaxf(rm, __shfl_xor(rm, 8, 32));
      float mn = fmaxf(mrow[r], rm);
      float corr = expf(mrow[r] - mn);
      mrow[r] = mn;
      float pa = expf(a - mn);
      float pb = expf(cc - mn);
      float rs = pa + pb;
      rs += __shfl_xor(rs, 1, 32);
      rs += __shfl_xor(rs, 2, 32);
      rs += __shfl_xor(rs, 4, 32);
      rs += __shfl_xor(rs, 8, 32);
      lrow[r] = lrow[r] * corr + rs;
#pragma unroll
      for (int nc = 0; nc < 8; ++nc) o[nc][r] *= corr;
      Pbuf[r + 8 * hi][ln] = pa;
      Pbuf[r + 8 * hi][ln + 16] = pb;
    }
    __syncthreads();
    // reload P as bf16 A-fragment (16x32)
    BF16x16 pfr;
#pragma unroll
    for (int j = 0; j < 8; ++j) {
      int ko = a_koff(j, hi);
      float p0 = Pbuf[ln][ko];
      float p1 = Pbuf[ln][ko + 1];
      pfr.u[j] = (unsigned)f2bf(p0) | ((unsigned)f2bf(p1) << 16);
    }
    __syncthreads();
    // O += P * V  (8 hd-subtiles)
#pragma unroll
    for (int nc = 0; nc < 8; ++nc) {
      BF16x16 vb;
#pragma unroll
      for (int j = 0; j < 8; ++j)
        vb.u[j] = *(const unsigned*)(vptr + (size_t)(nc * 16 + ln) * T_ + s0 +
                                     a_koff(j, hi));
      o[nc] = __builtin_amdgcn_wmma_f32_16x16x32_bf16(
          false, pfr.v, false, vb.v, (short)0, o[nc], false, false);
    }
  }
  // normalize and store O as bf16, layout (B*T, H*HD)
#pragma unroll
  for (int nc = 0; nc < 8; ++nc)
#pragma unroll
    for (int r = 0; r < 8; ++r) {
      int row = q0 + r + 8 * hi;
      float val = o[nc][r] / lrow[r];
      Obf[(size_t)(b * T_ + row) * (H_ * HD_) + h * HD_ + nc * 16 + ln] =
          f2bf(val);
    }
}

// ---------------- host launcher ----------------
extern "C" void kernel_launch(void* const* d_in, const int* in_sizes, int n_in,
                              void* d_out, int out_size, void* d_ws,
                              size_t ws_size, hipStream_t stream) {
  const float* x  = (const float*)d_in[0];
  const float* wq = (const float*)d_in[1];
  const float* wk = (const float*)d_in[2];
  const float* wv = (const float*)d_in[3];
  const float* wo = (const float*)d_in[4];
  const float* qg = (const float*)d_in[5];
  const float* kg = (const float*)d_in[6];

  const int M = B_ * T_;        // 4096
  const int NQ = H_ * HD_;      // 2048
  const int NKV = KVH_ * HD_;   // 512

  char* ws = (char*)d_ws;
  size_t off = 0;
  auto alloc = [&](size_t bytes) -> char* {
    char* p = ws + off;
    off += (bytes + 255) & ~(size_t)255;
    return p;
  };
  unsigned short* xbf = (unsigned short*)alloc((size_t)M * D_ * 2);
  unsigned short* wqb = (unsigned short*)alloc((size_t)NQ * D_ * 2);
  unsigned short* wkb = (unsigned short*)alloc((size_t)NKV * D_ * 2);
  unsigned short* wvb = (unsigned short*)alloc((size_t)NKV * D_ * 2);
  unsigned short* wob = (unsigned short*)alloc((size_t)D_ * NQ * 2);
  float* qf = (float*)alloc((size_t)M * NQ * 4);
  float* kf = (float*)alloc((size_t)M * NKV * 4);
  float* vf = (float*)alloc((size_t)M * NKV * 4);
  unsigned short* qn = (unsigned short*)alloc((size_t)M * NQ * 2);
  unsigned short* kn = (unsigned short*)alloc((size_t)M * NKV * 2);
  unsigned short* vt = (unsigned short*)alloc((size_t)M * NKV * 2);
  unsigned short* obf = (unsigned short*)alloc((size_t)M * NQ * 2);

  // 1) converts to bf16
  cvt_kernel<<<1024, 256, 0, stream>>>(x, xbf, M * D_);
  cvt_kernel<<<1024, 256, 0, stream>>>(wq, wqb, NQ * D_);
  cvt_kernel<<<512, 256, 0, stream>>>(wk, wkb, NKV * D_);
  cvt_kernel<<<512, 256, 0, stream>>>(wv, wvb, NKV * D_);
  cvt_kernel<<<1024, 256, 0, stream>>>(wo, wob, D_ * NQ);

  // 2) QKV projections (TDM-fed GEMM, 128x128 C tiles)
  gemm_bf16_kernel<<<dim3(NQ / 128, M / 128), 256, 0, stream>>>(xbf, wqb, qf,
                                                                M, NQ, D_);
  gemm_bf16_kernel<<<dim3(NKV / 128, M / 128), 256, 0, stream>>>(xbf, wkb, kf,
                                                                 M, NKV, D_);
  gemm_bf16_kernel<<<dim3(NKV / 128, M / 128), 256, 0, stream>>>(xbf, wvb, vf,
                                                                 M, NKV, D_);

  // 3) QK-norm (+ fold attention scale into Q), V transpose
  const int qrows = M * H_;
  const int krows = M * KVH_;
  rmsnorm_head_kernel<<<(qrows + 7) / 8, 256, 0, stream>>>(qf, qg, qn, H_,
                                                           SCALE_, qrows);
  rmsnorm_head_kernel<<<(krows + 7) / 8, 256, 0, stream>>>(kf, kg, kn, KVH_,
                                                           1.0f, krows);
  vtrans_kernel<<<1024, 256, 0, stream>>>(vf, vt);

  // 4) attention
  attn_kernel<<<dim3(T_ / 16, H_, B_), 32, 0, stream>>>(qn, kn, vt, obf);

  // 5) output projection -> d_out (f32)
  gemm_bf16_kernel<<<dim3(D_ / 128, M / 128), 256, 0, stream>>>(
      obf, wob, (float*)d_out, M, D_, NQ);
}
